// SlotAttentionSmolensky_45818711113931
// MI455X (gfx1250) — matrix operand order repaired and checked by reading, not compiled, not run on hardware
//
#include <hip/hip_runtime.h>
#include <hip/hip_bf16.h>

#define B_    32
#define N_    4096
#define FD_   256
#define D_    256
#define H_    4
#define NF_   8
#define DH_   64
#define ITERS_ 3

typedef __bf16 bf16_t;
typedef __attribute__((ext_vector_type(16))) __bf16 v16bf;
typedef __attribute__((ext_vector_type(8)))  __bf16 v8bf;
typedef __attribute__((ext_vector_type(8)))  float  v8f;
typedef __attribute__((ext_vector_type(4)))  unsigned int u32x4;
typedef __attribute__((ext_vector_type(8)))  int i32x8;
typedef __attribute__((ext_vector_type(4)))  int i32x4;

__device__ __forceinline__ v8f wmma_bf16(v16bf a, v16bf b, v8f c) {
  return __builtin_amdgcn_wmma_f32_16x16x32_bf16(false, a, false, b, (short)0, c, false, false);
}

// A-operand per-lane: two contiguous 8-element (16B) chunks along K
__device__ __forceinline__ v16bf load_a16(const bf16_t* p0, const bf16_t* p1) {
  v8bf lo = *(const v8bf*)p0;
  v8bf hi = *(const v8bf*)p1;
  return __builtin_shufflevector(lo, hi, 0,1,2,3,4,5,6,7,8,9,10,11,12,13,14,15);
}
// B-operand per-lane: 16 contiguous elements (32B) along K for one output column
__device__ __forceinline__ v16bf load_b16(const bf16_t* p) {
  return *(const v16bf*)p;
}

// ---- Tensor Data Mover: DMA a 16x256 bf16 tile (row stride 256) into LDS ----
// D# per CDNA5 ISA: group0 = {count, lds_addr, global_addr, type=2},
// group1 = {data_size=2B, tensor_dim0=256, tensor_dim1=131072, tile=[256,16], strides=256}.
__device__ __forceinline__ void tdm_load_16x256(unsigned lds_off, const bf16_t* gptr) {
  unsigned long long ga = (unsigned long long)(__UINTPTR_TYPE__)gptr;
  u32x4 g0;
  g0[0] = 1u;                                        // count=1, user mode
  g0[1] = lds_off;                                   // lds_addr (bytes)
  g0[2] = (unsigned)(ga & 0xffffffffu);              // global_addr[31:0]
  g0[3] = (unsigned)((ga >> 32) & 0x1ffffffu)        // global_addr[56:32]
        | (2u << 30);                                // type=2 ("image")
  i32x8 g1;
  g1[0] = (int)(1u << 16);                           // workgroup_mask=0, data_size=1 (2 bytes)
  g1[1] = (int)(256u << 16);                         // tensor_dim0=256 (low16 at bits 79:64... [31:16])
  g1[2] = 0;                                         // dim0 hi=0; tensor_dim1 low16 (131072 & 0xffff = 0)
  g1[3] = (int)(2u | (256u << 16));                  // tensor_dim1 hi=2; tile_dim0=256
  g1[4] = 16;                                        // tile_dim1=16; tile_dim2=0
  g1[5] = 256;                                       // tensor_dim0_stride low32
  g1[6] = (int)(256u << 16);                         // stride0 hi=0; tensor_dim1_stride low16
  g1[7] = 0;                                         // stride1 hi
  i32x4 gz = {0, 0, 0, 0};                           // groups 2/3 unused (2D tensor)
#if __clang_major__ >= 23
  i32x8 gz8 = {0, 0, 0, 0, 0, 0, 0, 0};
  __builtin_amdgcn_tensor_load_to_lds(g0, g1, gz, gz, gz8, 0);
#else
  __builtin_amdgcn_tensor_load_to_lds(g0, g1, gz, gz, 0);
#endif
}

// ---------------- LayerNorm(inputs) -> bf16, one wave32 per row of 256 ----------------
__global__ void sas_ln_in(const float* __restrict__ x, const float* __restrict__ g,
                          const float* __restrict__ bt, bf16_t* __restrict__ xn) {
  int wave = threadIdx.x >> 5;
  int lane = threadIdx.x & 31;
  long row = (long)blockIdx.x * 8 + wave;          // B*N rows
  const float* xr = x + row * FD_;
  float v[8]; float s = 0.f;
  #pragma unroll
  for (int e = 0; e < 8; ++e) { v[e] = xr[lane * 8 + e]; s += v[e]; }
  #pragma unroll
  for (int off = 16; off > 0; off >>= 1) s += __shfl_xor(s, off, 32);
  float mean = s * (1.f / FD_);
  float q = 0.f;
  #pragma unroll
  for (int e = 0; e < 8; ++e) { float d = v[e] - mean; q += d * d; }
  #pragma unroll
  for (int off = 16; off > 0; off >>= 1) q += __shfl_xor(q, off, 32);
  float rstd = rsqrtf(q * (1.f / FD_) + 1e-5f);
  bf16_t* o = xn + row * FD_;
  #pragma unroll
  for (int e = 0; e < 8; ++e) {
    int c = lane * 8 + e;
    o[c] = (bf16_t)(((v[e] - mean) * rstd) * g[c] + bt[c]);
  }
}

__global__ void sas_cvt_bf16(const float* __restrict__ a, bf16_t* __restrict__ o, int n) {
  int i = blockIdx.x * blockDim.x + threadIdx.x;
  if (i < n) o[i] = (bf16_t)a[i];
}
__global__ void sas_zero_bf16(bf16_t* __restrict__ o, int n) {
  int i = blockIdx.x * blockDim.x + threadIdx.x;
  if (i < n) o[i] = (bf16_t)0.f;
}

// ---------------- K/V projection: xn[BN,256] x Wk^T / Wv^T via WMMA bf16 ----------------
// block = 8 waves; waves 0-3 -> K quarters, waves 4-7 -> V quarters (stored transposed).
// A tile (16x256 bf16) staged once per block via TDM into LDS; B tiles double-buffered.
__global__ void sas_kv_gemm(const bf16_t* __restrict__ xn,
                            const bf16_t* __restrict__ wkb,  // [256][256] = Wk[d][f]
                            const bf16_t* __restrict__ wvb,
                            bf16_t* __restrict__ kout,       // [B*H][N][DH]
                            bf16_t* __restrict__ vtout) {    // [B*H][DH][N]
  __shared__ __align__(32) bf16_t sA[16 * FD_];    // 8 KB
  int wave = threadIdx.x >> 5;
  int lane = threadIdx.x & 31;
  long m0 = (long)blockIdx.x * 16;                 // tile of 16 rows of B*N
  int b  = (int)(m0 >> 12);
  int j0 = (int)(m0 & (N_ - 1));
  const bf16_t* wb = (wave < 4) ? wkb : wvb;
  int d0base = (wave & 3) * 64;
  int ah = lane >> 4, bcol = lane & 15, bhf = lane >> 4;

  if (wave == 0) {
    tdm_load_16x256((unsigned)(__UINTPTR_TYPE__)&sA[0], xn + m0 * FD_);
    __builtin_amdgcn_s_wait_tensorcnt(0);
  }
  __syncthreads();

  const bf16_t* arow = &sA[(lane & 15) * FD_];
  v16bf Bcur[4], Bnxt[4];
  #pragma unroll
  for (int t = 0; t < 4; ++t)
    Bcur[t] = load_b16(wb + (long)(d0base + t * 16 + bcol) * FD_ + bhf * 16);

  v8f acc[4] = {};
  for (int kk = 0; kk < 8; ++kk) {                 // K reduce = 256 in steps of 32
    int f0 = kk * 32 + ah * 16;
    v16bf A = load_a16(arow + f0, arow + f0 + 8);
    if (kk < 7) {
      int f = (kk + 1) * 32 + bhf * 16;
      #pragma unroll
      for (int t = 0; t < 4; ++t)
        Bnxt[t] = load_b16(wb + (long)(d0base + t * 16 + bcol) * FD_ + f);
    }
    #pragma unroll
    for (int t = 0; t < 4; ++t) acc[t] = wmma_bf16(A, Bcur[t], acc[t]);
    #pragma unroll
    for (int t = 0; t < 4; ++t) Bcur[t] = Bnxt[t];
  }
  #pragma unroll
  for (int t = 0; t < 4; ++t) {
    int d = d0base + t * 16 + (lane & 15);
    int h = d >> 6, dh = d & 63;
    long bh = (long)b * H_ + h;
    #pragma unroll
    for (int g = 0; g < 8; ++g) {
      int j = j0 + (lane >> 4) * 8 + g;            // C layout: M=(lane/16)*8+g, N=lane%16
      bf16_t val = (bf16_t)acc[t][g];
      if (wave < 4) kout[(bh * N_ + j) * DH_ + dh] = val;
      else          vtout[(bh * DH_ + dh) * N_ + j] = val;
    }
  }
}

// ---------------- LN(fillers) + Q projection (tiny, f32) -> bf16 q [B*H][16][64] ----------------
__global__ void sas_lnq(const float* __restrict__ fill, const float* __restrict__ g,
                        const float* __restrict__ bt, const float* __restrict__ Wq,
                        bf16_t* __restrict__ qout) {
  __shared__ float sf[D_];
  __shared__ float red[8];
  int row = blockIdx.x;                            // b*8+i, 256 rows
  int t = threadIdx.x;                             // 256
  float x = fill[(long)row * D_ + t];
  float s = x;
  #pragma unroll
  for (int off = 16; off > 0; off >>= 1) s += __shfl_xor(s, off, 32);
  if ((t & 31) == 0) red[t >> 5] = s;
  __syncthreads();
  float mean = 0.f;
  #pragma unroll
  for (int w = 0; w < 8; ++w) mean += red[w];
  mean *= (1.f / D_);
  float dx = x - mean;
  float q2 = dx * dx;
  #pragma unroll
  for (int off = 16; off > 0; off >>= 1) q2 += __shfl_xor(q2, off, 32);
  __syncthreads();
  if ((t & 31) == 0) red[t >> 5] = q2;
  __syncthreads();
  float var = 0.f;
  #pragma unroll
  for (int w = 0; w < 8; ++w) var += red[w];
  var *= (1.f / D_);
  sf[t] = dx * rsqrtf(var + 1e-5f) * g[t] + bt[t];
  __syncthreads();
  float acc = 0.f;
  const float* wr = Wq + (long)t * D_;
  for (int f = 0; f < D_; ++f) acc += sf[f] * wr[f];
  int b = row >> 3, i = row & 7;
  int h = t >> 6, dh = t & 63;
  qout[(((long)b * H_ + h) * 16 + i) * DH_ + dh] = (bf16_t)acc;
}

// ---------------- dots = Q x K^T via WMMA, one wave per (b,h,256-j chunk) ----------------
__global__ void sas_dots(const bf16_t* __restrict__ qb,   // [B*H][16][64] (rows 8..15 zero)
                         const bf16_t* __restrict__ kb,   // [B*H][N][64]
                         float* __restrict__ attn) {      // [B][32][N]
  int lane = threadIdx.x & 31;
  int id = blockIdx.x;
  int jchunk = id & 15; id >>= 4;
  int h = id & 3; int b = id >> 2;
  long bh = (long)b * H_ + h;
  const bf16_t* qp = qb + (bh * 16 + (lane & 15)) * DH_;
  int ah = (lane >> 4) * 16;
  v16bf A0 = load_a16(qp + ah,      qp + ah + 8);
  v16bf A1 = load_a16(qp + 32 + ah, qp + 32 + ah + 8);
  int jb = jchunk * 256;
  for (int jt = 0; jt < 16; ++jt) {
    int j0 = jb + jt * 16;
    const bf16_t* kp = kb + (bh * N_ + j0 + (lane & 15)) * DH_ + (lane >> 4) * 16;
    __builtin_prefetch(kp + 16 * DH_, 0, 0);
    v8f acc = {};
    acc = wmma_bf16(A0, load_b16(kp),      acc);
    acc = wmma_bf16(A1, load_b16(kp + 32), acc);
    if (lane < 16) {                               // rows M=0..7 live in lanes 0-15
      int j = j0 + lane;
      #pragma unroll
      for (int g = 0; g < 8; ++g)
        attn[((long)b * 32 + (g * H_ + h)) * N_ + j] = acc[g] * 0.125f;  // *DH^-0.5
    }
  }
}

// ---------------- inverted softmax over the 32 slot-head axis; emit attn_out on last iter ----------------
__global__ void sas_softmax(float* __restrict__ attn, float* __restrict__ attn_out, int write_out) {
  long idx = (long)blockIdx.x * blockDim.x + threadIdx.x;   // B*N threads
  int b = (int)(idx >> 12);
  int j = (int)(idx & (N_ - 1));
  float vals[32]; float mx = -1e30f;
  #pragma unroll
  for (int s = 0; s < 32; ++s) { float t = attn[((long)b * 32 + s) * N_ + j]; vals[s] = t; mx = fmaxf(mx, t); }
  float sum = 0.f;
  #pragma unroll
  for (int s = 0; s < 32; ++s) { float e = __expf(vals[s] - mx); vals[s] = e; sum += e; }
  float inv = 1.f / sum;
  #pragma unroll
  for (int s = 0; s < 32; ++s) { vals[s] *= inv; attn[((long)b * 32 + s) * N_ + j] = vals[s]; }
  if (write_out) {
    #pragma unroll
    for (int i = 0; i < NF_; ++i)
      attn_out[((long)b * NF_ + i) * N_ + j] =
          (vals[i*4] + vals[i*4+1] + vals[i*4+2] + vals[i*4+3]) * 0.25f;
  }
}

// ---------------- per (b,s) row sums over j (denominator incl EPS*N) ----------------
__global__ void sas_rowsum(const float* __restrict__ attn, float* __restrict__ sums) {
  __shared__ float red[8];
  int row = blockIdx.x;                            // B*32
  int t = threadIdx.x;
  const float* p = attn + (long)row * N_;
  float s = 0.f;
  for (int j = t; j < N_; j += 256) s += p[j];
  #pragma unroll
  for (int off = 16; off > 0; off >>= 1) s += __shfl_xor(s, off, 32);
  if ((t & 31) == 0) red[t >> 5] = s;
  __syncthreads();
  if (t == 0) {
    float tot = 0.f;
    #pragma unroll
    for (int w = 0; w < 8; ++w) tot += red[w];
    sums[row] = tot + 1e-8f * (float)N_;
  }
}

// ---------------- (attn+eps)/rowsum -> bf16 A-operand [B*H][16][N] (rows 8..15 stay zero) ----------------
__global__ void sas_anorm(const float* __restrict__ attn, const float* __restrict__ sums,
                          bf16_t* __restrict__ abf) {
  long idx = (long)blockIdx.x * blockDim.x + threadIdx.x;   // B*32*N
  int j = (int)(idx & (N_ - 1));
  long r = idx >> 12;                              // b*32 + s
  int s = (int)(r & 31);
  int b = (int)(r >> 5);
  int i = s >> 2, h = s & 3;
  float a = (attn[idx] + 1e-8f) / sums[r];
  abf[(((long)b * H_ + h) * 16 + i) * N_ + j] = (bf16_t)a;
}

// ---------------- updates = a x V via WMMA, split over 16 j-chunks, f32 partials ----------------
__global__ void sas_upd_gemm(const bf16_t* __restrict__ abf,  // [B*H][16][N]
                             const bf16_t* __restrict__ vt,   // [B*H][64][N]
                             float* __restrict__ upart) {     // [B*H][16ch][16][64]
  int lane = threadIdx.x & 31;
  int id = blockIdx.x;
  int ch = id & 15; long bh = id >> 4;             // bh in [0,128)
  int jb = ch * 256;
  const bf16_t* ap = abf + (bh * 16 + (lane & 15)) * N_;
  int ah = (lane >> 4) * 16, bcol = lane & 15, bhf = (lane >> 4) * 16;
  v8f acc[4] = {};
  for (int kk = 0; kk < 8; ++kk) {                 // 256 j per chunk, steps of 32
    int jA = jb + kk * 32 + ah;
    v16bf A = load_a16(ap + jA, ap + jA + 8);
    #pragma unroll
    for (int t = 0; t < 4; ++t) {
      int d = t * 16 + bcol;
      const bf16_t* vp = vt + (bh * 64 + d) * N_ + jb + kk * 32 + bhf;
      acc[t] = wmma_bf16(A, load_b16(vp), acc[t]);
    }
  }
  #pragma unroll
  for (int t = 0; t < 4; ++t) {
    int d = t * 16 + (lane & 15);
    #pragma unroll
    for (int g = 0; g < 8; ++g) {
      int i = (lane >> 4) * 8 + g;
      upart[((bh * 16 + ch) * 16 + i) * 64 + d] = acc[t][g];
    }
  }
}

__global__ void sas_upd_reduce(const float* __restrict__ upart, float* __restrict__ u) {
  int idx = blockIdx.x * blockDim.x + threadIdx.x;  // B*NF*D = 65536
  int col = idx & 255, row = idx >> 8;
  int b = row >> 3, i = row & 7;
  int h = col >> 6, dh = col & 63;
  long bh = (long)b * H_ + h;
  float s = 0.f;
  #pragma unroll
  for (int ch = 0; ch < 16; ++ch) s += upart[((bh * 16 + ch) * 16 + i) * 64 + dh];
  u[idx] = s;
}

// ---------------- GRU cell, f32, LDS-staged rows ----------------
__global__ void sas_gru(const float* __restrict__ u, const float* __restrict__ hprev,
                        const float* __restrict__ Wih, const float* __restrict__ Whh,
                        const float* __restrict__ bih, const float* __restrict__ bhh,
                        float* __restrict__ hout) {
  __shared__ float su[D_], sh[D_];
  int row = blockIdx.x, t = threadIdx.x;           // 256 rows x 256 cols
  su[t] = u[(long)row * D_ + t];
  sh[t] = hprev[(long)row * D_ + t];
  __syncthreads();
  float gi[3], gh[3];
  #pragma unroll
  for (int c = 0; c < 3; ++c) {
    const float* wi = Wih + (long)(c * D_ + t) * D_;
    const float* wh = Whh + (long)(c * D_ + t) * D_;
    float a = bih[c * D_ + t], b2 = bhh[c * D_ + t];
    for (int f = 0; f < D_; ++f) { a += su[f] * wi[f]; b2 += sh[f] * wh[f]; }
    gi[c] = a; gh[c] = b2;
  }
  float r = 1.f / (1.f + __expf(-(gi[0] + gh[0])));
  float z = 1.f / (1.f + __expf(-(gi[1] + gh[1])));
  float n = tanhf(gi[2] + r * gh[2]);
  hout[(long)row * D_ + t] = (1.f - z) * n + z * sh[t];
}

extern "C" void kernel_launch(void* const* d_in, const int* in_sizes, int n_in,
                              void* d_out, int out_size, void* d_ws, size_t ws_size,
                              hipStream_t stream) {
  (void)in_sizes; (void)n_in; (void)out_size; (void)ws_size;
  const float* inputs  = (const float*)d_in[0];
  const float* cond    = (const float*)d_in[1];
  const float* Wq      = (const float*)d_in[2];
  const float* Wk      = (const float*)d_in[3];
  const float* Wv      = (const float*)d_in[4];
  const float* Wih     = (const float*)d_in[5];
  const float* Whh     = (const float*)d_in[6];
  const float* bih     = (const float*)d_in[7];
  const float* bhh     = (const float*)d_in[8];
  const float* ln_in_g = (const float*)d_in[9];
  const float* ln_in_b = (const float*)d_in[10];
  const float* ln_f_g  = (const float*)d_in[11];
  const float* ln_f_b  = (const float*)d_in[12];

  float* slots_out = (float*)d_out;                          // [B,NF,D] = 65536
  float* attn_out  = (float*)d_out + (size_t)B_ * NF_ * D_;  // [B,NF,N]

  char* w = (char*)d_ws;
  auto carve = [&](size_t bytes) { char* p = w; w += (bytes + 255) & ~(size_t)255; return p; };
  bf16_t* xn   = (bf16_t*)carve((size_t)B_ * N_ * FD_ * sizeof(bf16_t));
  bf16_t* wkb  = (bf16_t*)carve((size_t)D_ * FD_ * sizeof(bf16_t));
  bf16_t* wvb  = (bf16_t*)carve((size_t)D_ * FD_ * sizeof(bf16_t));
  bf16_t* kb   = (bf16_t*)carve((size_t)B_ * H_ * N_ * DH_ * sizeof(bf16_t));
  bf16_t* vt   = (bf16_t*)carve((size_t)B_ * H_ * DH_ * N_ * sizeof(bf16_t));
  float*  attn = (float*) carve((size_t)B_ * 32 * N_ * sizeof(float));
  float*  sums = (float*) carve((size_t)B_ * 32 * sizeof(float));
  bf16_t* abf  = (bf16_t*)carve((size_t)B_ * H_ * 16 * N_ * sizeof(bf16_t));
  bf16_t* qb   = (bf16_t*)carve((size_t)B_ * H_ * 16 * DH_ * sizeof(bf16_t));
  float*  upart= (float*) carve((size_t)B_ * H_ * 16 * 16 * 64 * sizeof(float));
  float*  u    = (float*) carve((size_t)B_ * NF_ * D_ * sizeof(float));
  float*  fa   = (float*) carve((size_t)B_ * NF_ * D_ * sizeof(float));
  float*  fb   = (float*) carve((size_t)B_ * NF_ * D_ * sizeof(float));

  // weight conversion + pad-row zeroing (once per launch; deterministic)
  sas_cvt_bf16<<<(D_*FD_ + 255)/256, 256, 0, stream>>>(Wk, wkb, D_*FD_);
  sas_cvt_bf16<<<(D_*FD_ + 255)/256, 256, 0, stream>>>(Wv, wvb, D_*FD_);
  { int n = B_*H_*16*N_;  sas_zero_bf16<<<(n+255)/256, 256, 0, stream>>>(abf, n); }
  { int n = B_*H_*16*DH_; sas_zero_bf16<<<(n+255)/256, 256, 0, stream>>>(qb,  n); }

  // LN(inputs) -> bf16, then K/V projections (WMMA + TDM-staged A tiles)
  sas_ln_in<<<(B_*N_)/8, 256, 0, stream>>>(inputs, ln_in_g, ln_in_b, xn);
  sas_kv_gemm<<<(B_*N_)/16, 256, 0, stream>>>(xn, wkb, wvb, kb, vt);

  for (int it = 0; it < ITERS_; ++it) {
    const float* prev = (it == 0) ? cond : (it == 1 ? fa : fb);
    float* hnext = (it == ITERS_ - 1) ? slots_out : (it == 0 ? fa : fb);
    sas_lnq<<<B_*NF_, 256, 0, stream>>>(prev, ln_f_g, ln_f_b, Wq, qb);
    sas_dots<<<B_*H_*(N_/256), 32, 0, stream>>>(qb, kb, attn);
    sas_softmax<<<(B_*N_)/256, 256, 0, stream>>>(attn, attn_out, it == ITERS_ - 1);
    sas_rowsum<<<B_*32, 256, 0, stream>>>(attn, sums);
    sas_anorm<<<(B_*32*N_)/256, 256, 0, stream>>>(attn, sums, abf);
    sas_upd_gemm<<<B_*H_*16, 32, 0, stream>>>(abf, vt, upart);
    sas_upd_reduce<<<(B_*NF_*D_)/256, 256, 0, stream>>>(upart, u);
    sas_gru<<<B_*NF_, 256, 0, stream>>>(u, prev, Wih, Whh, bih, bhh, hnext);
  }
}
